// HDSharedExpertBasis_82497731821707
// MI455X (gfx1250) — matrix operand order repaired and checked by reading, not compiled, not run on hardware
//
#include <hip/hip_runtime.h>
#include <hip/hip_bf16.h>

typedef __attribute__((ext_vector_type(16))) __bf16 v16bf;
typedef __attribute__((ext_vector_type(8)))  __bf16 v8bf;
typedef __attribute__((ext_vector_type(8)))  float  v8f;

union Frag16 { v16bf v; v8bf h[2]; };

#define N_TOKENS 8192
#define D_MODEL  1024
#define HD_DIM   4096
#define N_VEC    32

// ---------------------------------------------------------------------------
// TDM support detection (per-pass: device pass sees the builtin, host doesn't)
// ---------------------------------------------------------------------------
#if defined(__has_builtin)
#if __has_builtin(__builtin_amdgcn_tensor_load_to_lds) && \
    __has_builtin(__builtin_amdgcn_s_wait_tensorcnt)
#define HAVE_TDM 1
#endif
#endif
#ifndef HAVE_TDM
#define HAVE_TDM 0
#endif

#if HAVE_TDM
typedef unsigned int u32x4 __attribute__((ext_vector_type(4)));
typedef int          i32x4 __attribute__((ext_vector_type(4)));
typedef int          i32x8 __attribute__((ext_vector_type(8)));

// Issue a TDM load of a 128-row x 64-element bf16 tile (row stride = K elems)
// into LDS at lds_byte_off, packed rows (pitch = 128 bytes).
// D# layout per CDNA5 ISA 8.3/8.4: group0 = {flags, lds_addr, gaddr lo/hi+type},
// group1 = {mask/data_size, dims, tile dims, dim0 stride}.  Tile always
// in-bounds, so tensor dims == tile dims.
__device__ __forceinline__ void tdm_load_tile_bf16(const void* gptr,
                                                   unsigned lds_byte_off,
                                                   int row_stride_elems) {
  unsigned long long ga = (unsigned long long)gptr;
  u32x4 g0;
  g0[0] = 1u;                                     // count=1 (valid user D#)
  g0[1] = lds_byte_off;                           // lds_addr
  g0[2] = (unsigned)ga;                           // global_addr[31:0]
  g0[3] = (unsigned)((ga >> 32) & 0x01FFFFFFu)    // global_addr[56:32]
        | (2u << 30);                             // type = 2 ("image")
  i32x8 g1;
  g1[0] = 1 << 16;                                // data_size = 1 (2 bytes)
  g1[1] = (64 & 0xFFFF) << 16;                    // tensor_dim0 = 64
  g1[2] = (128 & 0xFFFF) << 16;                   // tensor_dim1 = 128
  g1[3] = 64 << 16;                               // tile_dim0 = 64
  g1[4] = 128;                                    // tile_dim1 = 128, tile_dim2 = 0
  g1[5] = row_stride_elems;                       // tensor_dim0_stride[31:0]
  g1[6] = 0;                                      // stride hi / dim1_stride lo
  g1[7] = 0;
  i32x4 g2 = {0, 0, 0, 0};
  i32x4 g3 = {0, 0, 0, 0};
#if __clang_major__ >= 23
  i32x8 gx = {0, 0, 0, 0, 0, 0, 0, 0};
  __builtin_amdgcn_tensor_load_to_lds(g0, g1, g2, g3, gx, 0);
#else
  __builtin_amdgcn_tensor_load_to_lds(g0, g1, g2, g3, 0);
#endif
}
#endif  // HAVE_TDM

// ---------------------------------------------------------------------------
// inputs fp32 -> bf16 (vector of 4 per thread)
// ---------------------------------------------------------------------------
__global__ __launch_bounds__(256) void cvt_f32_to_bf16(
    const float* __restrict__ in, __bf16* __restrict__ out, int n4) {
  int i = blockIdx.x * blockDim.x + threadIdx.x;
  if (i >= n4) return;
  float4 v = ((const float4*)in)[i];
  __bf16 o[4] = {(__bf16)v.x, (__bf16)v.y, (__bf16)v.z, (__bf16)v.w};
  *(uint2*)(out + (size_t)i * 4) = *(uint2*)o;
}

// ---------------------------------------------------------------------------
// W^T[c][r] = sum_v coeffs[v] * basis[v][r][c]     (basis contiguous along c)
// Streams 537 MB contiguously; writes the 8 MB bf16 result transposed so the
// GEMM B-operand is [N][K] row-major (K contiguous).
// ---------------------------------------------------------------------------
__global__ __launch_bounds__(256) void reconstruct_bf16_T(
    const float* __restrict__ basis,       // [N_VEC][R][C]
    const float* __restrict__ coeffs_all,  // [N_EXP][N_VEC]
    const int* __restrict__ expert_idx,
    __bf16* __restrict__ outT,             // [C][R]
    int R, int C) {
  int t = blockIdx.x * blockDim.x + threadIdx.x;
  int c4perR = C >> 2;
  int total = R * c4perR;
  if (t >= total) return;
  int r = t / c4perR;
  int c = (t - r * c4perR) << 2;

  const float* coeffs = coeffs_all + expert_idx[0] * N_VEC;
  const size_t slab = (size_t)R * C;
  const float* p = basis + (size_t)r * C + c;

  float a0 = 0.f, a1 = 0.f, a2 = 0.f, a3 = 0.f;
#pragma unroll
  for (int v = 0; v < N_VEC; ++v) {
    float cv = coeffs[v];
    float4 b = *(const float4*)(p + (size_t)v * slab);
    __builtin_prefetch(p + (size_t)v * slab + 1024, 0, 1);
    a0 = fmaf(cv, b.x, a0);
    a1 = fmaf(cv, b.y, a1);
    a2 = fmaf(cv, b.z, a2);
    a3 = fmaf(cv, b.w, a3);
  }
  outT[(size_t)(c + 0) * R + r] = (__bf16)a0;
  outT[(size_t)(c + 1) * R + r] = (__bf16)a1;
  outT[(size_t)(c + 2) * R + r] = (__bf16)a2;
  outT[(size_t)(c + 3) * R + r] = (__bf16)a3;
}

// ---------------------------------------------------------------------------
// C[M][N] = A[M][K] @ Bt[N][K]^T  via v_wmma_f32_16x16x32_bf16.
// 256 threads = 8 waves (2x4). Block tile 128x128, BK=64.
// Tiles staged by the Tensor Data Mover (double buffered, TENSORcnt paced);
// fallback = cooperative b128 copies if the TDM builtin is unavailable.
// GELU_OUT: writes bf16 gelu(x); else writes f32 x + bias[n].
// ---------------------------------------------------------------------------
template <bool GELU_OUT>
__global__ __launch_bounds__(256) void wmma_gemm(
    const __bf16* __restrict__ A,   // [M][K]
    const __bf16* __restrict__ Bt,  // [N][K]
    __bf16* __restrict__ Obf,       // [M][N] if GELU_OUT
    float* __restrict__ Of,         // [M][N] if !GELU_OUT
    const float* __restrict__ bias_all,
    const int* __restrict__ expert_idx,
    int M, int N, int K) {
  constexpr int BK = 64;
  constexpr int TILE = 128 * BK;            // elements per tile (16 KB)
  extern __shared__ __bf16 sm[];            // [2][A-tile | B-tile], 64 KB

  const int tid  = threadIdx.x;
  const int lane = tid & 31;
  const int wid  = tid >> 5;
  const int waveM = wid >> 2;   // 0..1
  const int waveN = wid & 3;    // 0..3
  const int l15 = lane & 15;
  const int lhi = lane >> 4;

  const int m0 = blockIdx.y * 128;
  const int n0 = blockIdx.x * 128;
  const int T = K / BK;

  v8f acc[4][2] = {};

#if HAVE_TDM
  if (wid == 0) {
    tdm_load_tile_bf16(A  + (size_t)m0 * K, 0u,                    K);
    tdm_load_tile_bf16(Bt + (size_t)n0 * K, (unsigned)(TILE * 2),  K);
  }
#endif

  for (int t = 0; t < T; ++t) {
#if HAVE_TDM
    const int cur = t & 1;
    if (wid == 0) {
      if (t + 1 < T) {
        const int nxt = cur ^ 1;
        const size_t kn = (size_t)(t + 1) * BK;
        tdm_load_tile_bf16(A  + (size_t)m0 * K + kn,
                           (unsigned)(nxt * 4 * TILE), K);
        tdm_load_tile_bf16(Bt + (size_t)n0 * K + kn,
                           (unsigned)(nxt * 4 * TILE + 2 * TILE), K);
        __builtin_amdgcn_s_wait_tensorcnt(2);  // current buffer complete
      } else {
        __builtin_amdgcn_s_wait_tensorcnt(0);
      }
    }
#else
    const int cur = 0;
    // Cooperative staging: 16B chunks, A then B.
#pragma unroll
    for (int ch = tid; ch < TILE / 8; ch += 256) {
      int row = ch >> 3;
      int kc  = (ch & 7) << 3;
      *(uint4*)&sm[row * BK + kc] =
          *(const uint4*)&A[(size_t)(m0 + row) * K + t * BK + kc];
      *(uint4*)&sm[2 * TILE + row * BK + kc] =
          *(const uint4*)&Bt[(size_t)(n0 + row) * K + t * BK + kc];
    }
#endif
    __syncthreads();

    const __bf16* Abase = sm + cur * 4 * TILE;
    const __bf16* Bbase = Abase + 2 * TILE;

#pragma unroll
    for (int ks = 0; ks < BK; ks += 32) {
      Frag16 af[4], bfrag[2];
      // A fragment: lane half lhi holds K in [8*lhi,+8) and [16+8*lhi,+8).
#pragma unroll
      for (int mt = 0; mt < 4; ++mt) {
        const __bf16* ap = Abase + (waveM * 64 + mt * 16 + l15) * BK + ks + lhi * 8;
        af[mt].h[0] = *(const v8bf*)(ap);
        af[mt].h[1] = *(const v8bf*)(ap + 16);
      }
      // B fragment: lane half lhi holds K in [8*lhi,+8) and [16+8*lhi,+8) of column N=l15.
#pragma unroll
      for (int nt = 0; nt < 2; ++nt) {
        const __bf16* bp = Bbase + (waveN * 32 + nt * 16 + l15) * BK + ks + lhi * 8;
        bfrag[nt].h[0] = *(const v8bf*)(bp);
        bfrag[nt].h[1] = *(const v8bf*)(bp + 16);
      }
#pragma unroll
      for (int mt = 0; mt < 4; ++mt)
#pragma unroll
        for (int nt = 0; nt < 2; ++nt)
          acc[mt][nt] = __builtin_amdgcn_wmma_f32_16x16x32_bf16(
              false, af[mt].v, false, bfrag[nt].v, (short)0, acc[mt][nt],
              false, false);
    }
    __syncthreads();  // protect buffer before it is refilled
  }

  const float* bias = nullptr;
  if constexpr (!GELU_OUT) bias = bias_all + expert_idx[0] * D_MODEL;

  // C/D layout: VGPR r holds (M = r + 8*lhi, N = l15) of each 16x16 tile.
#pragma unroll
  for (int mt = 0; mt < 4; ++mt)
#pragma unroll
    for (int nt = 0; nt < 2; ++nt) {
      int gn = n0 + waveN * 32 + nt * 16 + l15;
#pragma unroll
      for (int r = 0; r < 8; ++r) {
        int gm = m0 + waveM * 64 + mt * 16 + lhi * 8 + r;
        float x = acc[mt][nt][r];
        if constexpr (GELU_OUT) {
          float g = 0.5f * x * (1.0f + erff(x * 0.70710678118654752f));
          Obf[(size_t)gm * N + gn] = (__bf16)g;
        } else {
          Of[(size_t)gm * N + gn] = x + bias[gn];
        }
      }
    }
}

// ---------------------------------------------------------------------------
extern "C" void kernel_launch(void* const* d_in, const int* in_sizes, int n_in,
                              void* d_out, int out_size, void* d_ws, size_t ws_size,
                              hipStream_t stream) {
  const float* inputs   = (const float*)d_in[0];
  const float* basis_up = (const float*)d_in[1];
  const float* basis_dn = (const float*)d_in[2];
  const float* cup      = (const float*)d_in[3];
  const float* cdn      = (const float*)d_in[4];
  const float* bias     = (const float*)d_in[5];
  const int*   eidx     = (const int*)d_in[6];
  float* out = (float*)d_out;

  char* ws = (char*)d_ws;
  __bf16* Xb   = (__bf16*)(ws);                                // 16 MB
  __bf16* WupT = (__bf16*)(ws + (size_t)16 * 1024 * 1024);     //  8 MB [h][d]
  __bf16* WdnT = (__bf16*)(ws + (size_t)24 * 1024 * 1024);     //  8 MB [d][h]
  __bf16* Hid  = (__bf16*)(ws + (size_t)32 * 1024 * 1024);     // 64 MB

  const size_t lds_bytes = 65536;  // 2 x (16 KB A-tile + 16 KB B-tile)

  // inputs -> bf16
  cvt_f32_to_bf16<<<(N_TOKENS * D_MODEL / 4) / 256, 256, 0, stream>>>(
      inputs, Xb, N_TOKENS * D_MODEL / 4);

  // W_up^T[h][d], W_down^T[d][h] from the shared bases (bandwidth-bound)
  reconstruct_bf16_T<<<(D_MODEL * HD_DIM / 4) / 256, 256, 0, stream>>>(
      basis_up, cup, eidx, WupT, D_MODEL, HD_DIM);
  reconstruct_bf16_T<<<(HD_DIM * D_MODEL / 4) / 256, 256, 0, stream>>>(
      basis_dn, cdn, eidx, WdnT, HD_DIM, D_MODEL);

  // hidden = gelu(X @ W_up)          (bf16 out)
  wmma_gemm<true><<<dim3(HD_DIM / 128, N_TOKENS / 128), 256, lds_bytes, stream>>>(
      Xb, WupT, Hid, nullptr, nullptr, eidx, N_TOKENS, HD_DIM, D_MODEL);

  // out = hidden @ W_down + bias     (f32 out)
  wmma_gemm<false><<<dim3(D_MODEL / 128, N_TOKENS / 128), 256, lds_bytes, stream>>>(
      Hid, WdnT, nullptr, out, bias, eidx, N_TOKENS, D_MODEL, HD_DIM);
}
